// Transformer_69475390980733
// MI455X (gfx1250) — matrix operand order repaired and checked by reading, not compiled, not run on hardware
//
#include <hip/hip_runtime.h>
#include <hip/hip_bf16.h>
#include <cstdint>
#include <cstddef>

typedef _Float16 v16h __attribute__((ext_vector_type(16)));
typedef _Float16 v8h  __attribute__((ext_vector_type(8)));
typedef float    v8f  __attribute__((ext_vector_type(8)));

union V16 { v16h v; v8h h[2]; };

#define BATCH 4
#define FLEN  4096
#define TLEN  512
#define DIMC  512
#define HEADS 8
#define HDIM  64

static __device__ __forceinline__ v8f wmma_f16(const V16& a, const V16& b, v8f c) {
  return __builtin_amdgcn_wmma_f32_16x16x32_f16(false, a.v, false, b.v, (short)0, c, false, false);
}

// ---------- weight transpose + f32->f16 : Wt[n][k] = W[k][n] ----------
__global__ __launch_bounds__(256) void k_transpose_w(const float* __restrict__ W,
                                                     _Float16* __restrict__ Wt,
                                                     int K, int N) {
  int idx = blockIdx.x * blockDim.x + threadIdx.x;   // over N*K
  if (idx >= N * K) return;
  int n = idx / K, k = idx - n * K;
  Wt[(size_t)n * K + k] = (_Float16)W[(size_t)k * N + n];
}

// ---------- WMMA GEMM: C[M,N] = A(f32)[M,K] @ Bt(f16)[N,K]^T + bias + addend[m%addMod] ----------
// block = 256 threads (8 waves), tile 128x64; each wave owns a 32x32 region = 2x2 WMMA tiles.
// Per K-step(32): 2 A-frags + 2 B-frags -> 4 v_wmma (each fragment reused twice).
__global__ __launch_bounds__(256) void k_gemm16(const float* __restrict__ A,
                                                const _Float16* __restrict__ Bt,
                                                const float* __restrict__ bias,
                                                const float* __restrict__ addend, int addMod,
                                                float* __restrict__ Cf,
                                                _Float16* __restrict__ Ch,
                                                int N, int K) {
  __shared__ _Float16 As[128][32];   // 8 KB
  __shared__ _Float16 Bs[64][32];    // 4 KB
  const int mblk = blockIdx.x * 128;
  const int nblk = blockIdx.y * 64;
  const int tid  = threadIdx.x;
  const int w    = tid >> 5, lane = tid & 31;
  const int wm   = w & 3, wn = w >> 2;              // wave grid 4x2 over (128,64)
  const int l15  = lane & 15;
  const int koffA = (lane >> 4) * 8;                // A-frag: halves {koffA..+7, koffA+16..+23}
  const int koffB = (lane >> 4) * 8;                // B-frag: halves {koffB..+7, koffB+16..+23}
  // staging coords
  const int sAr = tid >> 1;            // 0..127
  const int sAc = (tid & 1) * 16;      // 0 / 16
  const int sBr = tid >> 2;            // 0..63
  const int sBc = (tid & 3) * 8;       // 0,8,16,24
  v8f acc00 = {}, acc01 = {}, acc10 = {}, acc11 = {};
  for (int k0 = 0; k0 < K; k0 += 32) {
    { // stage A tile 128x32, convert f32->f16: 16 elems/thread
      const float* src = A + (size_t)(mblk + sAr) * K + k0 + sAc;
#pragma unroll
      for (int j = 0; j < 16; ++j) As[sAr][sAc + j] = (_Float16)src[j];
      if (k0 + 32 < K) __builtin_prefetch(src + 32, 0, 1);      // global_prefetch_b8 next tile
    }
    { // stage Bt tile 64x32 (f16): 8 halves (16B) per thread
      const _Float16* src = Bt + (size_t)(nblk + sBr) * K + k0 + sBc;
      *(v8h*)&Bs[sBr][sBc] = *(const v8h*)src;
      if (k0 + 32 < K) __builtin_prefetch(src + 32, 0, 1);
    }
    __syncthreads();
    V16 a0, a1, b0, b1;
    const int ar0 = wm * 32 + l15, ar1 = ar0 + 16;
    const int br0 = wn * 32 + l15, br1 = br0 + 16;
    a0.h[0] = *(const v8h*)&As[ar0][koffA];
    a0.h[1] = *(const v8h*)&As[ar0][koffA + 16];
    a1.h[0] = *(const v8h*)&As[ar1][koffA];
    a1.h[1] = *(const v8h*)&As[ar1][koffA + 16];
    b0.h[0] = *(const v8h*)&Bs[br0][koffB];
    b0.h[1] = *(const v8h*)&Bs[br0][koffB + 16];
    b1.h[0] = *(const v8h*)&Bs[br1][koffB];
    b1.h[1] = *(const v8h*)&Bs[br1][koffB + 16];
    acc00 = wmma_f16(a0, b0, acc00);
    acc01 = wmma_f16(a0, b1, acc01);
    acc10 = wmma_f16(a1, b0, acc10);
    acc11 = wmma_f16(a1, b1, acc11);
    __syncthreads();
  }
  // epilogue: 4 C tiles per wave
  const int mw = mblk + wm * 32;
  const int nw = nblk + wn * 32;
  const int mhalf = 8 * (lane >> 4);
  v8f accs[4] = {acc00, acc01, acc10, acc11};
#pragma unroll
  for (int tile = 0; tile < 4; ++tile) {
    const int mt = tile >> 1, ntile = tile & 1;     // acc[mt][ntile]
    const int n = nw + ntile * 16 + l15;
    const int mbase = mw + mt * 16 + mhalf;
    const float bv = bias ? bias[n] : 0.f;
    v8f acc = accs[tile];
#pragma unroll
    for (int r = 0; r < 8; ++r) {
      int m = mbase + r;
      float val = acc[r] + bv;
      if (addend) val += addend[(size_t)(m % addMod) * N + n];
      if (Cf) Cf[(size_t)m * N + n] = val;
      if (Ch) Ch[(size_t)m * N + n] = (_Float16)val;
    }
  }
}

// ---------- per-head V transpose: vT[((b*H+h)*HD+d)*T + t] = v16[(b*T+t)*DIM + h*HD + d] ----------
__global__ __launch_bounds__(256) void k_transpose_v(const _Float16* __restrict__ v,
                                                     _Float16* __restrict__ vT) {
  int idx = blockIdx.x * blockDim.x + threadIdx.x;   // B*H*HD*T = 1M
  int t = idx & (TLEN - 1);
  int rest = idx >> 9;
  int d = rest & (HDIM - 1);
  int bh = rest >> 6;
  int b = bh >> 3, h = bh & 7;
  vT[idx] = v[((size_t)(b * TLEN + t)) * DIMC + h * HDIM + d];
}

// ---------- gates: sigmoid(X@Wg + bg + P[r%posMod]@Wgp + bgp + extra), N=8 ----------
__global__ __launch_bounds__(256) void k_gate(const float* __restrict__ X,
                                              const float* __restrict__ P,
                                              const float* __restrict__ Wg,
                                              const float* __restrict__ bg,
                                              const float* __restrict__ Wgp,
                                              const float* __restrict__ bgp,
                                              float* __restrict__ out,
                                              int rows, int posMod, float extra) {
  int idx = blockIdx.x * blockDim.x + threadIdx.x;
  int r = idx >> 3, h = idx & 7;
  if (r >= rows) return;
  const float* x = X + (size_t)r * DIMC;
  const float* p = P + (size_t)(r % posMod) * DIMC;
  float acc = bg[h] + bgp[h] + extra;
  for (int k = 0; k < DIMC; ++k)
    acc += x[k] * Wg[k * HEADS + h] + p[k] * Wgp[k * HEADS + h];
  out[idx] = 1.f / (1.f + __expf(-acc));
}

// ---------- fused attention: scores(WMMA) -> softmax -> gates -> probs out -> PV(WMMA) ----------
// grid: (FLEN/16, BATCH*HEADS); block 256 (8 waves). Each block owns 16 f-rows of one head.
__global__ __launch_bounds__(256) void k_attn(const _Float16* __restrict__ q,
                                              const _Float16* __restrict__ kk16,
                                              const _Float16* __restrict__ vT,
                                              const float* __restrict__ g_to,
                                              const float* __restrict__ g_from,
                                              float* __restrict__ probs,
                                              float* __restrict__ control) {
  __shared__ float    sS[16][516];   // raw scores (padded)
  __shared__ _Float16 sP[16][520];   // f16 probs for PV (padded, 16B-aligned rows)
  __shared__ float    sCtrl[16][64]; // cross-wave PV reduction
  const int fbase = blockIdx.x * 16;
  const int b = blockIdx.y >> 3, h = blockIdx.y & 7;
  const int w = threadIdx.x >> 5, lane = threadIdx.x & 31;
  const int l15 = lane & 15;
  const int koffA = (lane >> 4) * 8;
  const int koffB = (lane >> 4) * 8;
  const int mrow  = 8 * (lane >> 4);

  // ---- phase 1: scores = q @ k^T / 8 ; each wave handles 4 of 32 t-tiles
  {
    const _Float16* qr = q + ((size_t)(b * FLEN + fbase + l15)) * DIMC + h * HDIM;
    V16 a0, a1;
    a0.h[0] = *(const v8h*)(qr + koffA);
    a0.h[1] = *(const v8h*)(qr + koffA + 16);
    a1.h[0] = *(const v8h*)(qr + 32 + koffA);
    a1.h[1] = *(const v8h*)(qr + 32 + koffA + 16);
    const _Float16* kb = kk16 + ((size_t)(b * TLEN)) * DIMC + h * HDIM;
    for (int ct = w * 4; ct < w * 4 + 4; ++ct) {
      const int tbase = ct * 16;
      const _Float16* kr = kb + (size_t)(tbase + l15) * DIMC;
      V16 b0, b1;
      b0.h[0] = *(const v8h*)(kr + koffB);
      b0.h[1] = *(const v8h*)(kr + koffB + 16);
      b1.h[0] = *(const v8h*)(kr + 32 + koffB);
      b1.h[1] = *(const v8h*)(kr + 32 + koffB + 16);
      v8f acc = {};
      acc = wmma_f16(a0, b0, acc);
      acc = wmma_f16(a1, b1, acc);
#pragma unroll
      for (int r = 0; r < 8; ++r)
        sS[mrow + r][tbase + l15] = acc[r] * 0.125f;   // 1/sqrt(64)
    }
  }
  __syncthreads();

  // ---- phase 2: softmax over T=512 per row, apply gates, emit probs (f32 global + f16 LDS)
  {
    const int row = threadIdx.x >> 4;
    const int g = threadIdx.x & 15;
    const float gf = g_from[((size_t)(b * FLEN + fbase + row)) * HEADS + h];
    float mx = -3.4e38f;
    for (int c = g; c < TLEN; c += 16) mx = fmaxf(mx, sS[row][c]);
#pragma unroll
    for (int m = 8; m; m >>= 1) mx = fmaxf(mx, __shfl_xor(mx, m, 16));
    float sum = 0.f;
    for (int c = g; c < TLEN; c += 16) sum += __expf(sS[row][c] - mx);
#pragma unroll
    for (int m = 8; m; m >>= 1) sum += __shfl_xor(sum, m, 16);
    const float inv = 1.f / sum;
    const size_t prow = ((size_t)(b * HEADS + h) * FLEN + fbase + row) * TLEN;
    for (int c = g; c < TLEN; c += 16) {
      float pp = __expf(sS[row][c] - mx) * inv
               * g_to[(size_t)(b * TLEN + c) * HEADS + h] * gf;
      probs[prow + c] = pp;
      sP[row][c] = (_Float16)pp;
    }
  }
  __syncthreads();

  // ---- phase 3: control = probs @ v ; waves split (4 n-tiles) x (2 K-halves)
  {
    const int nt = w & 3, kh = w >> 2;
    const _Float16* vb = vT + ((size_t)(b * HEADS + h) * HDIM + nt * 16 + l15) * TLEN;
    v8f cacc = {};
    for (int kc = 0; kc < 8; ++kc) {
      const int k0 = kh * 256 + kc * 32;
      V16 a, bm;
      a.h[0]  = *(const v8h*)&sP[l15][k0 + koffA];
      a.h[1]  = *(const v8h*)&sP[l15][k0 + koffA + 16];
      bm.h[0] = *(const v8h*)(vb + k0 + koffB);
      bm.h[1] = *(const v8h*)(vb + k0 + koffB + 16);
      cacc = wmma_f16(a, bm, cacc);
    }
    if (w >= 4) {
#pragma unroll
      for (int r = 0; r < 8; ++r) sCtrl[mrow + r][nt * 16 + l15] = cacc[r];
    }
    __syncthreads();
    if (w < 4) {
#pragma unroll
      for (int r = 0; r < 8; ++r) {
        float val = cacc[r] + sCtrl[mrow + r][nt * 16 + l15];
        control[((size_t)(b * FLEN + fbase + mrow + r)) * DIMC + h * HDIM + nt * 16 + l15] = val;
      }
    }
  }
}

// ---------- layernorm row stats of from_tensor ----------
__global__ __launch_bounds__(256) void k_rowstats(const float* __restrict__ X,
                                                  float* __restrict__ mu,
                                                  float* __restrict__ rstd,
                                                  int rows) {
  int r = blockIdx.x * blockDim.x + threadIdx.x;
  if (r >= rows) return;
  const float* x = X + (size_t)r * DIMC;
  float s = 0.f, ss = 0.f;
  for (int k = 0; k < DIMC; ++k) { float v = x[k]; s += v; ss += v * v; }
  float m = s * (1.f / DIMC);
  float var = ss * (1.f / DIMC) - m * m;
  mu[r] = m;
  rstd[r] = rsqrtf(var + 1e-5f);
}

// ---------- out = layernorm(f2) * (gain + 1) ----------
__global__ __launch_bounds__(256) void k_final(const float* __restrict__ f2,
                                               const float* __restrict__ mu,
                                               const float* __restrict__ rstd,
                                               const float* __restrict__ gain,
                                               float* __restrict__ out) {
  size_t idx = (size_t)blockIdx.x * blockDim.x + threadIdx.x;
  int r = (int)(idx >> 9);
  float tn = (f2[idx] - mu[r]) * rstd[r];
  out[idx] = tn * (gain[idx] + 1.f);
}

extern "C" void kernel_launch(void* const* d_in, const int* in_sizes, int n_in,
                              void* d_out, int out_size, void* d_ws, size_t ws_size,
                              hipStream_t stream) {
  (void)in_sizes; (void)n_in; (void)out_size; (void)ws_size;
  const float* from_tensor = (const float*)d_in[0];   // [B, F, 512]
  const float* to_tensor   = (const float*)d_in[1];   // [B, T, 512]
  const float* from_pos    = (const float*)d_in[2];   // [F, 512]
  const float* to_pos      = (const float*)d_in[3];   // [T, 512]
  const float* Wq  = (const float*)d_in[4];  const float* bq  = (const float*)d_in[5];
  const float* Wk  = (const float*)d_in[6];  const float* bk  = (const float*)d_in[7];
  const float* Wv  = (const float*)d_in[8];  const float* bv  = (const float*)d_in[9];
  const float* Wfp = (const float*)d_in[10]; const float* bfp = (const float*)d_in[11];
  const float* Wtp = (const float*)d_in[12]; const float* btp = (const float*)d_in[13];
  const float* Wg_to   = (const float*)d_in[14]; const float* bg_to   = (const float*)d_in[15];
  const float* Wgp_to  = (const float*)d_in[16]; const float* bgp_to  = (const float*)d_in[17];
  const float* Wg_from = (const float*)d_in[18]; const float* bg_from = (const float*)d_in[19];
  const float* Wgp_from= (const float*)d_in[20]; const float* bgp_from= (const float*)d_in[21];
  const float* Wm  = (const float*)d_in[22]; const float* bm  = (const float*)d_in[23];

  float* outp   = (float*)d_out;                               // [B*F, 512]
  float* probsp = outp + (size_t)BATCH * FLEN * DIMC;          // [B, H, F, T]

  // workspace carve-out (256B aligned)
  size_t off = 0;
  char* wsb = (char*)d_ws;
  auto alloc = [&](size_t bytes) -> void* {
    void* p = wsb + off;
    off += (bytes + 255) & ~(size_t)255;
    return p;
  };
  const size_t WB = (size_t)DIMC * DIMC;
  _Float16* WqT  = (_Float16*)alloc(WB * 2);
  _Float16* WkT  = (_Float16*)alloc(WB * 2);
  _Float16* WvT  = (_Float16*)alloc(WB * 2);
  _Float16* WfpT = (_Float16*)alloc(WB * 2);
  _Float16* WtpT = (_Float16*)alloc(WB * 2);
  _Float16* WmT  = (_Float16*)alloc(WB * 2);
  float*    fpW  = (float*)   alloc((size_t)FLEN * DIMC * 4);
  float*    tpW  = (float*)   alloc((size_t)TLEN * DIMC * 4);
  _Float16* q16  = (_Float16*)alloc((size_t)BATCH * FLEN * DIMC * 2);
  _Float16* k16  = (_Float16*)alloc((size_t)BATCH * TLEN * DIMC * 2);
  _Float16* v16  = (_Float16*)alloc((size_t)BATCH * TLEN * DIMC * 2);
  _Float16* vT16 = (_Float16*)alloc((size_t)BATCH * TLEN * DIMC * 2);
  float*    gto  = (float*)   alloc((size_t)BATCH * TLEN * HEADS * 4);
  float*    gfrom= (float*)   alloc((size_t)BATCH * FLEN * HEADS * 4);
  float*    ctrl = (float*)   alloc((size_t)BATCH * FLEN * DIMC * 4);
  float*    gain = (float*)   alloc((size_t)BATCH * FLEN * DIMC * 4);
  float*    mu   = (float*)   alloc((size_t)BATCH * FLEN * 4);
  float*    rstd = (float*)   alloc((size_t)BATCH * FLEN * 4);

  const int MROWS_F = BATCH * FLEN;   // 16384
  const int MROWS_T = BATCH * TLEN;   // 2048

  // 1) weight transposes (f32 -> f16 [N][K])
  {
    int nb = (DIMC * DIMC + 255) / 256;
    k_transpose_w<<<nb, 256, 0, stream>>>(Wq,  WqT,  DIMC, DIMC);
    k_transpose_w<<<nb, 256, 0, stream>>>(Wk,  WkT,  DIMC, DIMC);
    k_transpose_w<<<nb, 256, 0, stream>>>(Wv,  WvT,  DIMC, DIMC);
    k_transpose_w<<<nb, 256, 0, stream>>>(Wfp, WfpT, DIMC, DIMC);
    k_transpose_w<<<nb, 256, 0, stream>>>(Wtp, WtpT, DIMC, DIMC);
    k_transpose_w<<<nb, 256, 0, stream>>>(Wm,  WmT,  DIMC, DIMC);
  }

  // 2) position maps: fpW = from_pos@Wfp+bfp ; tpW = to_pos@Wtp+btp
  k_gemm16<<<dim3(FLEN / 128, DIMC / 64), 256, 0, stream>>>(
      from_pos, WfpT, bfp, nullptr, 1, fpW, nullptr, DIMC, DIMC);
  k_gemm16<<<dim3(TLEN / 128, DIMC / 64), 256, 0, stream>>>(
      to_pos, WtpT, btp, nullptr, 1, tpW, nullptr, DIMC, DIMC);

  // 3) q/k/v projections (f16 outputs; q,k get pos-map addends)
  k_gemm16<<<dim3(MROWS_F / 128, DIMC / 64), 256, 0, stream>>>(
      from_tensor, WqT, bq, fpW, FLEN, nullptr, q16, DIMC, DIMC);
  k_gemm16<<<dim3(MROWS_T / 128, DIMC / 64), 256, 0, stream>>>(
      to_tensor, WkT, bk, tpW, TLEN, nullptr, k16, DIMC, DIMC);
  k_gemm16<<<dim3(MROWS_T / 128, DIMC / 64), 256, 0, stream>>>(
      to_tensor, WvT, bv, nullptr, 1, nullptr, v16, DIMC, DIMC);

  // 4) per-head V transpose for PV WMMA B-operand
  k_transpose_v<<<(BATCH * HEADS * HDIM * TLEN) / 256, 256, 0, stream>>>(v16, vT16);

  // 5) gates
  k_gate<<<(MROWS_T * HEADS + 255) / 256, 256, 0, stream>>>(
      to_tensor, to_pos, Wg_to, bg_to, Wgp_to, bgp_to, gto, MROWS_T, TLEN, 0.f);
  k_gate<<<(MROWS_F * HEADS + 255) / 256, 256, 0, stream>>>(
      from_tensor, from_pos, Wg_from, bg_from, Wgp_from, bgp_from, gfrom, MROWS_F, FLEN, 1.0f);

  // 6) fused attention (scores -> softmax -> gates -> probs out -> PV)
  k_attn<<<dim3(FLEN / 16, BATCH * HEADS), 256, 0, stream>>>(
      q16, k16, vT16, gto, gfrom, probsp, ctrl);

  // 7) modulation GEMM: gain = control @ Wm + bm
  k_gemm16<<<dim3(MROWS_F / 128, DIMC / 64), 256, 0, stream>>>(
      ctrl, WmT, bm, nullptr, 1, gain, nullptr, DIMC, DIMC);

  // 8) layernorm stats + final elementwise
  k_rowstats<<<(MROWS_F + 255) / 256, 256, 0, stream>>>(from_tensor, mu, rstd, MROWS_F);
  k_final<<<(MROWS_F * DIMC) / 256, 256, 0, stream>>>(from_tensor, mu, rstd, gain, outp);
}
